// SelfAttention_4509715661123
// MI455X (gfx1250) — matrix operand order repaired and checked by reading, not compiled, not run on hardware
//
#include <hip/hip_runtime.h>

#define BATCH 2
#define HEADS 16
#define SEQ   2048
#define DH    64
#define QT    16
#define KT    64
#define THREADS 128
#define SCALE 0.125f
#define MASK_VAL -1e9f

typedef _Float16 v8h  __attribute__((ext_vector_type(8)));
typedef _Float16 v16h __attribute__((ext_vector_type(16)));
typedef float    v8f  __attribute__((ext_vector_type(8)));
typedef int      v4i  __attribute__((ext_vector_type(4)));

typedef __attribute__((address_space(1))) v4i gv4i;   // global int4
typedef __attribute__((address_space(3))) v4i lv4i;   // LDS int4

#if defined(__has_builtin)
#if __has_builtin(__builtin_amdgcn_global_load_async_to_lds_b128)
#define HAVE_ASYNC_LDS 1
#endif
#endif

// LDS partition offsets (bytes)
#define OFF_SQ   0
#define OFF_SKF  (OFF_SQ  + QT * DH * 2)        // 2048   (K chunk kept as f32)
#define OFF_SVT  (OFF_SKF + KT * DH * 4)        // +16384 (V^T as f16)
#define OFF_SS   (OFF_SVT + DH * KT * 2)        // +8192
#define OFF_MAX  (OFF_SS  + QT * SEQ * 4)       // +131072
#define OFF_SUM  (OFF_MAX + QT * 4)
#define SMEM_BYTES (OFF_SUM + QT * 4)

extern __shared__ char smem_raw[];

__device__ __forceinline__ v16h cat8h(v8h lo, v8h hi) {
    return __builtin_shufflevector(lo, hi, 0,1,2,3,4,5,6,7,8,9,10,11,12,13,14,15);
}

// convert 16 contiguous f32 (in LDS) to a v16h B-fragment
__device__ __forceinline__ v16h cvt16(const float* p) {
    v8f a = *(const v8f*)p;
    v8f b = *(const v8f*)(p + 8);
    v16h r;
    #pragma unroll
    for (int e = 0; e < 8; ++e) { r[e] = (_Float16)a[e]; r[8 + e] = (_Float16)b[e]; }
    return r;
}

// wave-local reduction within groups of 8 lanes via ds_swizzle SWAPX1/2/4
#define SWZ_MAXF(x, imm) \
    x = fmaxf(x, __int_as_float(__builtin_amdgcn_ds_swizzle(__float_as_int(x), imm)))
#define SWZ_ADDF(x, imm) \
    x = x + __int_as_float(__builtin_amdgcn_ds_swizzle(__float_as_int(x), imm))

__global__ __launch_bounds__(THREADS)
void fa_fwd_wmma(const float* __restrict__ Q, const float* __restrict__ K,
                 const float* __restrict__ V, float* __restrict__ Out,
                 float* __restrict__ Prob) {
    _Float16* sQ  = (_Float16*)(smem_raw + OFF_SQ);   // [QT][DH]  f16
    float*    sKf = (float*)   (smem_raw + OFF_SKF);  // [KT][DH]  f32 (async-staged)
    _Float16* sVt = (_Float16*)(smem_raw + OFF_SVT);  // [DH][KT]  f16 (V transposed)
    float*    sS  = (float*)   (smem_raw + OFF_SS);   // [QT][SEQ] scores -> probs
    float*    sMax= (float*)   (smem_raw + OFF_MAX);  // [QT]
    float*    sInv= (float*)   (smem_raw + OFF_SUM);  // [QT] 1/sum

    const int tid  = threadIdx.x;
    const int wave = tid >> 5;
    const int lane = tid & 31;
    const int hi   = lane >> 4;   // half-wave select
    const int ln   = lane & 15;

    const int bh = blockIdx.x / (SEQ / QT);
    const int q0 = (blockIdx.x % (SEQ / QT)) * QT;

    const float* gQ = Q + ((size_t)bh * SEQ + q0) * DH;
    const float* gK = K + (size_t)bh * SEQ * DH;
    const float* gV = V + (size_t)bh * SEQ * DH;
    float*       gO = Out  + ((size_t)bh * SEQ + q0) * DH;
    float*       gP = Prob + ((size_t)bh * SEQ + q0) * (size_t)SEQ;

    // causal: this tile only attends keys j < jc (rounded up to KT chunk)
    const int jc = min(SEQ, ((q0 + QT + KT - 1) / KT) * KT);

    // ---- load Q tile, fold in softmax scale, convert to f16 ----
    {
        const int base = tid * 8;                       // 128 thr * 8 = 1024 = 16*64
        float4 a = *(const float4*)(gQ + base);
        float4 b = *(const float4*)(gQ + base + 4);
        _Float16* d = sQ + base;
        d[0] = (_Float16)(a.x * SCALE); d[1] = (_Float16)(a.y * SCALE);
        d[2] = (_Float16)(a.z * SCALE); d[3] = (_Float16)(a.w * SCALE);
        d[4] = (_Float16)(b.x * SCALE); d[5] = (_Float16)(b.y * SCALE);
        d[6] = (_Float16)(b.z * SCALE); d[7] = (_Float16)(b.w * SCALE);
    }
    __syncthreads();

    // ---- build Q A-fragments once (A: 16x32 f16; lane ln = row M, split K by hi) ----
    v16h aQ0, aQ1;
    {
        const _Float16* row = sQ + ln * DH;
        aQ0 = cat8h(*(const v8h*)(row +  0 + hi * 16), *(const v8h*)(row +  8 + hi * 16));
        aQ1 = cat8h(*(const v8h*)(row + 32 + hi * 16), *(const v8h*)(row + 40 + hi * 16));
    }

    // ================= Phase 1: S = (Q*scale) @ K^T =================
    for (int kb = 0; kb < jc; kb += KT) {
        { // stage raw f32 K chunk [kb..kb+63][0..63] into LDS
            const int r  = tid >> 1;
            const int c0 = (tid & 1) * 32;
            const float* src = gK + (size_t)(kb + r) * DH + c0;
            float*       dst = sKf + r * DH + c0;
#ifdef HAVE_ASYNC_LDS
            #pragma unroll
            for (int e = 0; e < 8; ++e)     // 8 x b128 per lane, straight to LDS
                __builtin_amdgcn_global_load_async_to_lds_b128(
                    (gv4i*)(src + e * 4), (lv4i*)(dst + e * 4), 0, 0);
#if __has_builtin(__builtin_amdgcn_s_wait_asynccnt)
            __builtin_amdgcn_s_wait_asynccnt(0);
#else
            asm volatile("s_wait_asynccnt 0" ::: "memory");
#endif
#else
            #pragma unroll
            for (int e = 0; e < 32; e += 4)
                *(float4*)(dst + e) = *(const float4*)(src + e);
#endif
        }
        __syncthreads();

        const int jt = kb + wave * 16;           // each wave owns one 16-col tile
        if (jt <= q0 + QT - 1) {                 // skip fully-masked tiles
            const float* brow = sKf + (wave * 16 + ln) * DH;   // N=ln -> key row
            v16h b0 = cvt16(brow +  0 + hi * 16);              // K = d 0..31
            v16h b1 = cvt16(brow + 32 + hi * 16);              // K = d 32..63
            v8f c = {};
            c = __builtin_amdgcn_wmma_f32_16x16x32_f16(false, aQ0, false, b0,
                                                       (short)0, c, false, false);
            c = __builtin_amdgcn_wmma_f32_16x16x32_f16(false, aQ1, false, b1,
                                                       (short)0, c, false, false);
            #pragma unroll
            for (int r = 0; r < 8; ++r)          // C: M = r + 8*hi, N = ln
                sS[(r + hi * 8) * SEQ + jt + ln] = c[r];
        }
        __syncthreads();
    }

    // ================= softmax reductions (8 threads per row, float4 reads) ======
    {
        const int row = tid >> 3, sub = tid & 7;
        const int ig  = q0 + row;
        const float* srow = sS + row * SEQ;
        float m = MASK_VAL;
        for (int j = sub * 4; j < jc; j += 32) {
            float4 s = *(const float4*)(srow + j);
            s.x = (j + 0 <= ig) ? s.x : MASK_VAL;
            s.y = (j + 1 <= ig) ? s.y : MASK_VAL;
            s.z = (j + 2 <= ig) ? s.z : MASK_VAL;
            s.w = (j + 3 <= ig) ? s.w : MASK_VAL;
            m = fmaxf(m, fmaxf(fmaxf(s.x, s.y), fmaxf(s.z, s.w)));
        }
        SWZ_MAXF(m, 0x041F);   // SWAPX1
        SWZ_MAXF(m, 0x081F);   // SWAPX2
        SWZ_MAXF(m, 0x101F);   // SWAPX4
        float l = 0.f;
        for (int j = sub * 4; j < jc; j += 32) {
            float4 s = *(const float4*)(srow + j);
            l += (j + 0 <= ig) ? __expf(s.x - m) : 0.f;
            l += (j + 1 <= ig) ? __expf(s.y - m) : 0.f;
            l += (j + 2 <= ig) ? __expf(s.z - m) : 0.f;
            l += (j + 3 <= ig) ? __expf(s.w - m) : 0.f;
        }
        SWZ_ADDF(l, 0x041F);
        SWZ_ADDF(l, 0x081F);
        SWZ_ADDF(l, 0x101F);
        if (sub == 0) { sMax[row] = m; sInv[row] = 1.0f / l; }
    }
    __syncthreads();

    // ---- normalize in LDS + write probs to HBM (b128), zero-fill masked tail ----
    for (int i = 0; i < QT; ++i) {
        const float m = sMax[i], invl = sInv[i];
        const int ig = q0 + i;
        float* srow = sS + i * SEQ;
        float* prow = gP + (size_t)i * SEQ;
        for (int j = tid * 4; j < jc; j += THREADS * 4) {
            float4 s = *(const float4*)(srow + j);
            float4 p;
            p.x = (j + 0 <= ig) ? __expf(s.x - m) * invl : 0.f;
            p.y = (j + 1 <= ig) ? __expf(s.y - m) * invl : 0.f;
            p.z = (j + 2 <= ig) ? __expf(s.z - m) * invl : 0.f;
            p.w = (j + 3 <= ig) ? __expf(s.w - m) * invl : 0.f;
            *(float4*)(srow + j) = p;
            *(float4*)(prow + j) = p;
        }
        const float4 z4 = {0.f, 0.f, 0.f, 0.f};
        for (int j = jc + tid * 4; j < SEQ; j += THREADS * 4)
            *(float4*)(prow + j) = z4;
    }
    __syncthreads();

    // ================= Phase 2: O = P @ V =================
    v8f o = {};
    const int dtile = wave * 16;                 // each wave owns one 16-wide d tile
    for (int kb = 0; kb < jc; kb += KT) {
        { // stage V chunk transposed: sVt[d][jlocal] (f16)
            const int r  = tid >> 1;
            const int c0 = (tid & 1) * 32;
            const float* src = gV + (size_t)(kb + r) * DH + c0;
            #pragma unroll
            for (int e = 0; e < 32; e += 4) {
                float4 f = *(const float4*)(src + e);
                sVt[(c0+e+0) * KT + r] = (_Float16)f.x;
                sVt[(c0+e+1) * KT + r] = (_Float16)f.y;
                sVt[(c0+e+2) * KT + r] = (_Float16)f.z;
                sVt[(c0+e+3) * KT + r] = (_Float16)f.w;
            }
        }
        __syncthreads();

        #pragma unroll
        for (int sub = 0; sub < 2; ++sub) {
            const int j0 = sub * 32;
            // A: P tile 16x32, row M=ln, K=j; read f32 probs, convert to f16
            const float* pr = sS + ln * SEQ + kb + j0;
            v8f plo = *(const v8f*)(pr +     hi * 16);
            v8f phi = *(const v8f*)(pr + 8 + hi * 16);
            v16h a;
            #pragma unroll
            for (int e = 0; e < 8; ++e) { a[e] = (_Float16)plo[e]; a[8+e] = (_Float16)phi[e]; }
            // B: V tile 32x16, N=d (=ln within tile), K=j contiguous in sVt
            v16h b = *(const v16h*)(sVt + (dtile + ln) * KT + j0 + hi * 16);
            o = __builtin_amdgcn_wmma_f32_16x16x32_f16(false, a, false, b,
                                                       (short)0, o, false, false);
        }
        __syncthreads();
    }

    #pragma unroll
    for (int r = 0; r < 8; ++r)                  // D: M = r + 8*hi, N = ln
        gO[(r + hi * 8) * DH + dtile + ln] = o[r];
}

extern "C" void kernel_launch(void* const* d_in, const int* in_sizes, int n_in,
                              void* d_out, int out_size, void* d_ws, size_t ws_size,
                              hipStream_t stream) {
    const float* q = (const float*)d_in[0];
    const float* k = (const float*)d_in[1];
    const float* v = (const float*)d_in[2];
    // d_in[3] is the causal mask; causality is computed analytically in-kernel.

    float* outO = (float*)d_out;
    float* outP = outO + (size_t)BATCH * HEADS * SEQ * DH;

    (void)hipFuncSetAttribute((const void*)fa_fwd_wmma,
                              hipFuncAttributeMaxDynamicSharedMemorySize, SMEM_BYTES);

    const int grid = BATCH * HEADS * (SEQ / QT);
    fa_fwd_wmma<<<dim3(grid), dim3(THREADS), SMEM_BYTES, stream>>>(q, k, v, outO, outP);
}
